// CostVolumeAttention2_66546223284232
// MI455X (gfx1250) — matrix operand order repaired and checked by reading, not compiled, not run on hardware
//
#include <hip/hip_runtime.h>
#include <hip/hip_bf16.h>

typedef __attribute__((ext_vector_type(16))) _Float16 v16h;
typedef __attribute__((ext_vector_type(8)))  float    v8f;
typedef __attribute__((ext_vector_type(4)))  unsigned int u32x4;
typedef __attribute__((ext_vector_type(8)))  int i32x8;
typedef __attribute__((ext_vector_type(4)))  int i32x4;

union Frag16 { v16h v; uint4 u[2]; };
union Half8  { _Float16 h[8]; uint4 u; };

#define WMMA_F32_F16(A, B, C) \
  __builtin_amdgcn_wmma_f32_16x16x32_f16(false, (A), false, (B), (short)0, (C), false, false)

#if __has_builtin(__builtin_amdgcn_tensor_load_to_lds) && __has_builtin(__builtin_amdgcn_s_wait_tensorcnt)
#define USE_TDM 1
#else
#define USE_TDM 0
#endif

#if USE_TDM
// ---------------------------------------------------------------------------
// 1-D f16 tile copy global -> LDS via the Tensor Data Mover.
// n_elems contiguous 16-bit elements; optional LDS padding of 4 DWORDs after
// every 32 DWORDs (matches a [rows][64+8] half layout for 64-half rows).
// D# packing per CDNA5 ISA.
// This toolchain declares the 6-arg builtin:
//   (u32x4 g0, i32x8 g1, i32x4 g2, i32x4 g3, i32x8 extra, i32 cpol)
// ---------------------------------------------------------------------------
__device__ __forceinline__ void tdm_load_u16(const void* gsrc, void* ldst,
                                             unsigned n_elems, bool pad32dw)
{
    const unsigned lds = (unsigned)(size_t)ldst;              // LDS aperture: low 32 bits
    const unsigned long long ga = (unsigned long long)(size_t)gsrc;
    u32x4 g0;
    g0[0] = 1u;                                               // count=1 (valid), user mode
    g0[1] = lds;                                              // lds_addr       [63:32]
    g0[2] = (unsigned)(ga & 0xFFFFFFFFull);                   // global_addr lo [95:64]
    g0[3] = (unsigned)((ga >> 32) & 0x1FFFFFFull) | (2u << 30); // ga hi + type=2

    i32x8 g1 = {0, 0, 0, 0, 0, 0, 0, 0};
    unsigned w0 = (1u << 16);                                 // data_size = 1 (2 bytes)
    if (pad32dw) w0 |= (1u << 20)                             // pad_enable
                     | (4u << 22)                             // pad_interval: 32 DWORDs
                     | (3u << 25);                            // pad_amount:   4 DWORDs
    g1[0] = (int)w0;
    g1[1] = (int)((n_elems & 0xFFFFu) << 16);                 // tensor_dim0 [15:0] -> bits 63:48
    g1[2] = (int)(((n_elems >> 16) & 0xFFFFu) | (1u << 16));  // tensor_dim0 hi | tensor_dim1=1
    g1[3] = (int)((n_elems & 0xFFFFu) << 16);                 // tile_dim0 -> bits 127:112
    g1[4] = 1;                                                // tile_dim1 = 1
    g1[5] = (int)n_elems;                                     // tensor_dim0_stride lo
    const i32x4 z4 = {0, 0, 0, 0};
    const i32x8 z8 = {0, 0, 0, 0, 0, 0, 0, 0};
    __builtin_amdgcn_tensor_load_to_lds(g0, g1, z4, z4, z8, 0);
}
#endif

// ---------------------------------------------------------------------------
// GEMM: out[M x N] = X[M x K] * W[N x K]^T (+bias), f32 in/out, f16 WMMA core.
// Workgroup tile: 128 (M) x 64 (N). 8 waves, each wave: 16 rows x 64 cols.
// ---------------------------------------------------------------------------
__global__ __launch_bounds__(256)
void gemm_xwt_f16(const float* __restrict__ X, const float* __restrict__ W,
                  const float* __restrict__ bias, float* __restrict__ out,
                  int K, int ldout)
{
    __shared__ _Float16 As[128][40];   // [m][k], pad 8 halfs
    __shared__ _Float16 Bs[64][40];    // [n][k], pad 8 halfs

    const int tid  = threadIdx.x;
    const int lane = tid & 31;
    const int wv   = tid >> 5;
    const int half = lane >> 4;
    const int col  = lane & 15;
    const int m0   = blockIdx.y * 128;
    const int n0   = blockIdx.x * 64;

    v8f acc[4] = {};

    for (int kb = 0; kb < K; kb += 32) {
        __syncthreads();
        {   // A tile: 128x32 f32 -> f16, 16 elems/thread (contiguous)
            int e0 = tid * 16, r = e0 >> 5, c = e0 & 31;
            const float* s = X + (size_t)(m0 + r) * K + kb + c;
            Half8 h0, h1;
            #pragma unroll
            for (int i = 0; i < 8; i++) h0.h[i] = (_Float16)s[i];
            #pragma unroll
            for (int i = 0; i < 8; i++) h1.h[i] = (_Float16)s[8 + i];
            *(uint4*)&As[r][c]     = h0.u;
            *(uint4*)&As[r][c + 8] = h1.u;
        }
        {   // B tile: 64x32 f32 -> f16, 8 elems/thread
            int e0 = tid * 8, r = e0 >> 5, c = e0 & 31;
            const float* s = W + (size_t)(n0 + r) * K + kb + c;
            Half8 h0;
            #pragma unroll
            for (int i = 0; i < 8; i++) h0.h[i] = (_Float16)s[i];
            *(uint4*)&Bs[r][c] = h0.u;
        }
        __syncthreads();

        // gather all fragments first, then drain into back-to-back WMMAs
        Frag16 a, bfr[4];
        const int mrow = wv * 16 + col;
        a.u[0] = *(const uint4*)&As[mrow][half * 8];
        a.u[1] = *(const uint4*)&As[mrow][16 + half * 8];
        #pragma unroll
        for (int f = 0; f < 4; f++) {
            const int nc = f * 16 + col;
            bfr[f].u[0] = *(const uint4*)&Bs[nc][half * 8];
            bfr[f].u[1] = *(const uint4*)&Bs[nc][16 + half * 8];
        }
        #pragma unroll
        for (int f = 0; f < 4; f++)
            acc[f] = WMMA_F32_F16(a.v, bfr[f].v, acc[f]);
    }

    #pragma unroll
    for (int f = 0; f < 4; f++) {
        const int c = n0 + f * 16 + col;
        const float bi = bias ? bias[c] : 0.0f;
        #pragma unroll
        for (int r = 0; r < 8; r++) {
            const int row = m0 + wv * 16 + r + 8 * half;
            out[(size_t)row * ldout + c] = acc[f][r] + bi;
        }
    }
}

// ---------------------------------------------------------------------------
// L2-normalize q and frame-shifted k across hd=64 channels; emit f16,
// channel-last layout [g=128][pix=1024][64].  One wave per pixel.
// ---------------------------------------------------------------------------
__global__ __launch_bounds__(256)
void norm_qk(const float* __restrict__ qkv, _Float16* __restrict__ qf,
             _Float16* __restrict__ kf)
{
    const int tid  = threadIdx.x;
    const int lane = tid & 31;
    const int wv   = tid >> 5;
    const int p = blockIdx.x * 8 + wv;     // 0 .. 131071
    const int g = p >> 10, i = p & 1023;
    const int b = g >> 6, h = (g >> 3) & 7, t = g & 7;

    const size_t qrow = ((size_t)b * 8192 + (size_t)t * 1024 + i) * 1536 + h * 64;
    float q0 = qkv[qrow + lane], q1 = qkv[qrow + 32 + lane];
    float ss = q0 * q0 + q1 * q1;
    ss += __shfl_xor(ss, 1, 32);  ss += __shfl_xor(ss, 2, 32);
    ss += __shfl_xor(ss, 4, 32);  ss += __shfl_xor(ss, 8, 32);
    ss += __shfl_xor(ss, 16, 32);
    const float invq = 1.0f / fmaxf(sqrtf(ss), 1e-12f);
    const size_t o = ((size_t)g * 1024 + i) * 64;
    qf[o + lane]      = (_Float16)(q0 * invq);
    qf[o + 32 + lane] = (_Float16)(q1 * invq);

    const int tp = (t < 7) ? t + 1 : 7;    // k_post frame shift
    const size_t krow = ((size_t)b * 8192 + (size_t)tp * 1024 + i) * 1536 + 512 + h * 64;
    float k0 = qkv[krow + lane], k1 = qkv[krow + 32 + lane];
    float ks = k0 * k0 + k1 * k1;
    ks += __shfl_xor(ks, 1, 32);  ks += __shfl_xor(ks, 2, 32);
    ks += __shfl_xor(ks, 4, 32);  ks += __shfl_xor(ks, 8, 32);
    ks += __shfl_xor(ks, 16, 32);
    const float invk = 1.0f / fmaxf(sqrtf(ks), 1e-12f);
    kf[o + lane]      = (_Float16)(k0 * invk);
    kf[o + 32 + lane] = (_Float16)(k1 * invk);
}

// ---------------------------------------------------------------------------
// 7x7 windowed correlation. One WG per (slice g, image row y).
// corr stored f16 [g][pixel][u] with u padded 49 -> 64 (zeros).
// ---------------------------------------------------------------------------
__global__ __launch_bounds__(256)
void corr_win(const _Float16* __restrict__ qf, const _Float16* __restrict__ kf,
              _Float16* __restrict__ corr)
{
    __shared__ _Float16 qs[32][64];        // this row of qf (contiguous copy)
    __shared__ _Float16 ks[7][38][64];     // 7 rows x (32+6) cols, zero-padded

    const int tid = threadIdx.x;
    const int g = blockIdx.x, y = blockIdx.y;

#if USE_TDM
    if (tid < 32)   // wave 0 issues the DMA; other waves do the halo loads
        tdm_load_u16(qf + ((size_t)g * 1024 + (size_t)y * 32) * 64, &qs[0][0],
                     2048u, false);
#else
    {   // load q row: 2048 halfs
        int xr = tid >> 3, c0 = (tid & 7) * 8;
        *(uint4*)&qs[xr][c0] =
            *(const uint4*)(qf + ((size_t)g * 1024 + (size_t)y * 32 + xr) * 64 + c0);
    }
#endif
    for (int idx = tid; idx < 7 * 38; idx += 256) {
        int ry = idx / 38, xx = idx % 38;
        int yy = y + ry - 3, xg = xx - 3;
        uint4* d = (uint4*)&ks[ry][xx][0];
        if (yy >= 0 && yy < 32 && xg >= 0 && xg < 32) {
            const uint4* s = (const uint4*)(kf + ((size_t)g * 1024 + (size_t)yy * 32 + xg) * 64);
            #pragma unroll
            for (int e = 0; e < 8; e++) d[e] = s[e];
        } else {
            uint4 z; z.x = z.y = z.z = z.w = 0u;
            #pragma unroll
            for (int e = 0; e < 8; e++) d[e] = z;
        }
    }
#if USE_TDM
    if (tid < 32) __builtin_amdgcn_s_wait_tensorcnt(0);
#endif
    __syncthreads();

    for (int o = tid; o < 2048; o += 256) {
        const int xloc = o >> 6, u = o & 63;
        float r = 0.0f;
        if (u < 49) {
            const int dy = u / 7, dx = u % 7;
            const _Float16* qp = &qs[xloc][0];
            const _Float16* kp = &ks[dy][xloc + dx][0];
            #pragma unroll
            for (int d2 = 0; d2 < 64; d2++) r += (float)qp[d2] * (float)kp[d2];
            r *= (1.0f / 64.0f);
        }
        corr[((size_t)g * 1024 + (size_t)y * 32 + xloc) * 64 + u] = (_Float16)r;
    }
}

// ---------------------------------------------------------------------------
// Fused flash attention per slice g:
//   S = corr_i . corr_j^T  (K = 64 padded u), online softmax, out = P . V.
// Grid (128 slices, 8 i-blocks of 128 rows). 8 waves x 16 rows each.
// corr tiles arrive via TDM (with LDS row padding); v converts through VGPRs.
// ---------------------------------------------------------------------------
__global__ __launch_bounds__(256)
void flash_attn(const float* __restrict__ qkv, const _Float16* __restrict__ corr,
                float* __restrict__ attnout)
{
    __shared__ _Float16 cI[128][72];       // corr rows for this i-block  [i][u]
    __shared__ _Float16 cJ[32][72];        // corr rows for j-chunk       [j][u]
    __shared__ _Float16 vT[64][40];        // v chunk transposed          [d][j]
    __shared__ _Float16 pst[8][16][40];    // per-wave P staging          [m][j]

    const int tid  = threadIdx.x;
    const int lane = tid & 31;
    const int wv   = tid >> 5;
    const int half = lane >> 4;
    const int col  = lane & 15;
    const int g  = blockIdx.x;
    const int i0 = blockIdx.y * 128;
    const int b = g >> 6, h = (g >> 3) & 7, t = g & 7;
    const size_t rowbase = (size_t)b * 8192 + (size_t)t * 1024;
    const _Float16* cbase = corr + (size_t)g * 1024 * 64;

#if USE_TDM
    if (wv == 0)   // 128x64 halfs, contiguous source, 72-half LDS pitch via pad
        tdm_load_u16(cbase + (size_t)i0 * 64, &cI[0][0], 8192u, true);
#else
    {   // load corr i-block: 128 x 64 halfs
        int r = tid >> 1, c0 = (tid & 1) * 32;
        const uint4* s = (const uint4*)(cbase + (size_t)(i0 + r) * 64 + c0);
        uint4* d = (uint4*)&cI[r][c0];
        d[0] = s[0]; d[1] = s[1]; d[2] = s[2]; d[3] = s[3];
    }
#endif

    float m_run[8], l_run[8];
    #pragma unroll
    for (int r = 0; r < 8; r++) { m_run[r] = -3.0e38f; l_run[r] = 0.0f; }
    v8f acc[4] = {};

    for (int j0 = 0; j0 < 1024; j0 += 32) {
        __syncthreads();
#if USE_TDM
        if (wv == 0)   // corr j-chunk DMA overlaps the v conversion below
            tdm_load_u16(cbase + (size_t)j0 * 64, &cJ[0][0], 2048u, true);
#else
        {   // corr j-chunk: 32 x 64 halfs
            int r = tid >> 3, c0 = (tid & 7) * 8;
            *(uint4*)&cJ[r][c0] = *(const uint4*)(cbase + (size_t)(j0 + r) * 64 + c0);
        }
#endif
        {   // v chunk (32 x 64 f32 from qkv) -> vT[d][j] f16
            int j = tid >> 3, d0 = (tid & 7) * 8;
            const float* vs = qkv + (rowbase + j0 + j) * 1536 + 1024 + h * 64 + d0;
            #pragma unroll
            for (int e = 0; e < 8; e++) vT[d0 + e][j] = (_Float16)vs[e];
            if (j0 + 32 < 1024) __builtin_prefetch(vs + 32 * 1536, 0, 1);
        }
#if USE_TDM
        if (wv == 0) __builtin_amdgcn_s_wait_tensorcnt(0);
#endif
        __syncthreads();

        // ---- S = corr_i . corr_j^T : 16 rows x 32 cols per wave -----------
        const int mrow = wv * 16 + col;
        Frag16 a0, a1, b0, b1, b2, b3;
        a0.u[0] = *(const uint4*)&cI[mrow][half * 8];
        a0.u[1] = *(const uint4*)&cI[mrow][16 + half * 8];
        a1.u[0] = *(const uint4*)&cI[mrow][32 + half * 8];
        a1.u[1] = *(const uint4*)&cI[mrow][48 + half * 8];
        b0.u[0] = *(const uint4*)&cJ[col][half * 8];
        b0.u[1] = *(const uint4*)&cJ[col][16 + half * 8];
        b1.u[0] = *(const uint4*)&cJ[col][32 + half * 8];
        b1.u[1] = *(const uint4*)&cJ[col][48 + half * 8];
        b2.u[0] = *(const uint4*)&cJ[col + 16][half * 8];
        b2.u[1] = *(const uint4*)&cJ[col + 16][16 + half * 8];
        b3.u[0] = *(const uint4*)&cJ[col + 16][32 + half * 8];
        b3.u[1] = *(const uint4*)&cJ[col + 16][48 + half * 8];
        v8f s0 = {}, s1 = {};
        s0 = WMMA_F32_F16(a0.v, b0.v, s0);
        s0 = WMMA_F32_F16(a1.v, b1.v, s0);
        s1 = WMMA_F32_F16(a0.v, b2.v, s1);
        s1 = WMMA_F32_F16(a1.v, b3.v, s1);

        // ---- online softmax (rows live as r + 8*half across 16-lane halves)
        float p0[8], p1[8];
        #pragma unroll
        for (int r = 0; r < 8; r++) {
            float mx = fmaxf(s0[r], s1[r]);
            mx = fmaxf(mx, __shfl_xor(mx, 1, 16));
            mx = fmaxf(mx, __shfl_xor(mx, 2, 16));
            mx = fmaxf(mx, __shfl_xor(mx, 4, 16));
            mx = fmaxf(mx, __shfl_xor(mx, 8, 16));
            const float mnew  = fmaxf(m_run[r], mx);
            const float scale = __expf(m_run[r] - mnew);
            m_run[r] = mnew;
            p0[r] = __expf(s0[r] - mnew);
            p1[r] = __expf(s1[r] - mnew);
            float rs = p0[r] + p1[r];
            rs += __shfl_xor(rs, 1, 16);
            rs += __shfl_xor(rs, 2, 16);
            rs += __shfl_xor(rs, 4, 16);
            rs += __shfl_xor(rs, 8, 16);
            l_run[r] = l_run[r] * scale + rs;
            #pragma unroll
            for (int f = 0; f < 4; f++) acc[f][r] *= scale;
        }

        // ---- stage P (C-layout -> A-layout via LDS), then acc += P . V ----
        #pragma unroll
        for (int r = 0; r < 8; r++) {
            pst[wv][r + 8 * half][col]      = (_Float16)p0[r];
            pst[wv][r + 8 * half][col + 16] = (_Float16)p1[r];
        }
        __syncthreads();
        Frag16 pa, vb[4];
        pa.u[0] = *(const uint4*)&pst[wv][col][half * 8];
        pa.u[1] = *(const uint4*)&pst[wv][col][16 + half * 8];
        #pragma unroll
        for (int f = 0; f < 4; f++) {
            vb[f].u[0] = *(const uint4*)&vT[f * 16 + col][half * 8];
            vb[f].u[1] = *(const uint4*)&vT[f * 16 + col][16 + half * 8];
        }
        #pragma unroll
        for (int f = 0; f < 4; f++)
            acc[f] = WMMA_F32_F16(pa.v, vb[f].v, acc[f]);
    }

    // epilogue: normalize and scatter into (B, N, C) pre-projection buffer
    #pragma unroll
    for (int r = 0; r < 8; r++) {
        const float inv = 1.0f / l_run[r];
        #pragma unroll
        for (int f = 0; f < 4; f++) {
            const int row = i0 + wv * 16 + r + 8 * half;
            const int c   = h * 64 + f * 16 + col;
            attnout[(rowbase + row) * 512 + c] = acc[f][r] * inv;
        }
    }
}

// ---------------------------------------------------------------------------
extern "C" void kernel_launch(void* const* d_in, const int* in_sizes, int n_in,
                              void* d_out, int out_size, void* d_ws, size_t ws_size,
                              hipStream_t stream)
{
    (void)in_sizes; (void)n_in; (void)out_size; (void)ws_size;
    const float* x      = (const float*)d_in[0];   // [2,8192,512]
    const float* w_qkv  = (const float*)d_in[1];   // [1536,512]
    const float* w_proj = (const float*)d_in[2];   // [512,512]
    const float* b_proj = (const float*)d_in[3];   // [512]
    float* out = (float*)d_out;                    // [2,8192,512]

    char* ws = (char*)d_ws;
    float*    qkv     = (float*)ws;                          // 16384*1536*4 = 100,663,296 B
    _Float16* qf      = (_Float16*)(ws + 100663296ull);      // 128*1024*64*2 = 16,777,216 B
    _Float16* kf      = (_Float16*)(ws + 117440512ull);      // 16,777,216 B
    _Float16* corr    = (_Float16*)(ws + 134217728ull);      // 16,777,216 B (end 150,994,944)
    float*    attnout = (float*)(ws + 100663296ull);         // aliases qf+kf (dead by then)

    // 1) qkv = x @ w_qkv^T
    gemm_xwt_f16<<<dim3(24, 128), 256, 0, stream>>>(x, w_qkv, nullptr, qkv, 512, 1536);
    // 2) l2norm + frame shift -> qf, kf (f16)
    norm_qk<<<dim3(16384), 256, 0, stream>>>(qkv, qf, kf);
    // 3) windowed correlation -> corr (f16, u padded to 64)
    corr_win<<<dim3(128, 32), 256, 0, stream>>>(qf, kf, corr);
    // 4) fused corr^T corr -> softmax -> @ v  (flash style)
    flash_attn<<<dim3(128, 8), 256, 0, stream>>>(qkv, corr, attnout);
    // 5) final projection + bias
    gemm_xwt_f16<<<dim3(8, 128), 256, 0, stream>>>(attnout, w_proj, b_proj, out, 512, 512);
}
